// plasgraph_9191230013449
// MI455X (gfx1250) — matrix operand order repaired and checked by reading, not compiled, not run on hardware
//
#include <hip/hip_runtime.h>
#include <hip/hip_bf16.h>

typedef __attribute__((ext_vector_type(16))) __bf16 v16bf;
typedef __attribute__((ext_vector_type(8)))  float  v8f;

#if defined(__AMDGCN__) && __has_builtin(__builtin_amdgcn_global_load_async_to_lds_b128) && \
    __has_builtin(__builtin_amdgcn_s_wait_asynccnt)
#define USE_ASYNC_LDS 1
#else
#define USE_ASYNC_LDS 0
#endif

#if USE_ASYNC_LDS
typedef int nvi4 __attribute__((vector_size(4 * sizeof(int))));   // native int4
typedef __attribute__((address_space(1))) nvi4 gvec4_t;           // global
typedef __attribute__((address_space(3))) nvi4 lvec4_t;           // LDS
#endif

__device__ __forceinline__ unsigned short f2bf(float f) {
    unsigned int u = __float_as_uint(f);
    u += 0x7FFFu + ((u >> 16) & 1u);   // round-to-nearest-even
    return (unsigned short)(u >> 16);
}
__device__ __forceinline__ float bf2f(unsigned int u) {
    return __uint_as_float(u << 16);
}

// --------------------------------------------- weight cvt + transpose to [C][K]
// in: [layers][K][128] f32 row-major; out: [layers][128][K] bf16
__global__ void __launch_bounds__(256)
cvt_transpose_kernel(const float* __restrict__ in, unsigned short* __restrict__ out,
                     int K, int total) {
    int i = blockIdx.x * 256 + threadIdx.x;
    if (i < total) {
        int c = i & 127;
        int t = i >> 7;          // = l*K + k
        int k = t % K;
        int l = t / K;
        out[((size_t)l * 128 + c) * K + k] = f2bf(in[i]);   // coalesced read
    }
}

__global__ void __launch_bounds__(256)
zero_f32_kernel(float* __restrict__ p, int n) {
    int i = blockIdx.x * 256 + threadIdx.x;
    if (i < n) p[i] = 0.f;
}

// ------------------------------------------------- front: preproc + fc_in1/2
__global__ void __launch_bounds__(128)
front_kernel(const float* __restrict__ x,
             const float* __restrict__ Wp, const float* __restrict__ bp,
             const float* __restrict__ W1, const float* __restrict__ b1,
             const float* __restrict__ W2, const float* __restrict__ b2,
             unsigned short* __restrict__ node_id, unsigned short* __restrict__ h,
             int N)
{
    __shared__ float lWp[16 * 32];
    __shared__ float lbp[32];
    __shared__ float lW1[32 * 128];
    __shared__ float lb1[128];
    __shared__ float lW2[32 * 128];
    __shared__ float lb2[128];
    const int tid = threadIdx.x;
    for (int i = tid; i < 16 * 32; i += 128) lWp[i] = Wp[i];
    if (tid < 32) lbp[tid] = bp[tid];
    for (int i = tid; i < 32 * 128; i += 128) { lW1[i] = W1[i]; lW2[i] = W2[i]; }
    if (tid < 128) { lb1[tid] = b1[tid]; lb2[tid] = b2[tid]; }
    __syncthreads();

    const int n = blockIdx.x * 128 + tid;
    if (n < N) {
        float xl[16];
        #pragma unroll
        for (int i = 0; i < 16; ++i) xl[i] = x[(size_t)n * 16 + i];
        float hp[32];
        #pragma unroll
        for (int j = 0; j < 32; ++j) {
            float s = lbp[j];
            #pragma unroll
            for (int i = 0; i < 16; ++i) s += xl[i] * lWp[i * 32 + j];
            hp[j] = s > 0.f ? s : 0.f;
        }
        for (int j = 0; j < 128; ++j) {
            float s1 = lb1[j], s2 = lb2[j];
            #pragma unroll
            for (int i = 0; i < 32; ++i) {
                s1 += hp[i] * lW1[i * 128 + j];
                s2 += hp[i] * lW2[i * 128 + j];
            }
            node_id[(size_t)n * 128 + j] = f2bf(s1 > 0.f ? s1 : 0.f);
            h[(size_t)n * 128 + j]       = f2bf(s2 > 0.f ? s2 : 0.f);
        }
    } else {
        for (int j = 0; j < 128; ++j) {
            node_id[(size_t)n * 128 + j] = 0;
            h[(size_t)n * 128 + j]       = 0;
        }
    }
}

// ----------------------------------------------------------- WMMA GEMM C=128
// out[N,128] = act(concat(A0[:,0:K0], A1[:,0:Ktot-K0]) @ W[Ktot,128] + bias)
// A0/A1/out bf16 row stride 128; Wt bf16 PRE-TRANSPOSED [128][Ktot].
// Block: 256 threads = 8 waves; block tile 128 rows x 128 cols.
// Double-buffered LDS; async global->LDS staging overlaps chunk k+1 copy
// with chunk k WMMAs (one barrier + one s_wait_asynccnt per k-step).
__global__ void __launch_bounds__(256)
gemm128_kernel(const unsigned short* __restrict__ A0,
               const unsigned short* __restrict__ A1,
               const unsigned short* __restrict__ Wt,
               const float* __restrict__ bias,
               unsigned short* __restrict__ out,
               int N, int K0, int Ktot, int doRelu)
{
    __shared__ __align__(16) unsigned short lA[2][128 * 32];   // [row][k]
    __shared__ __align__(16) unsigned short lW[2][128 * 32];   // [col][k]
    __shared__ float lbias[128];

    const int tid  = threadIdx.x;
    const int lane = tid & 31;
    const int wave = tid >> 5;
    const int rowBase = blockIdx.x * 128;

    if (tid < 128) lbias[tid] = bias ? bias[tid] : 0.f;

    const int m    = lane & 15;     // M (A) / N (B,C) index within tile
    const int half = lane >> 4;
    const int ak   = half * 8;      // A frag: K = {ak..ak+7, ak+16..ak+23}
    const int bk   = half * 8;      // B frag: K = {bk..bk+7, bk+16..bk+23}
    const int rs   = tid >> 1;      // staging row/col (0..127)
    const int cs   = (tid & 1) * 16;// staging k offset (16 bf16 = 2x b128)

    const int nk = Ktot >> 5;
    v8f acc[8] = {};

    auto stage = [&](int kidx, int buf) {
        const int k0 = kidx * 32;
        const unsigned short* Aseg;
        int kloc;
        if (k0 < K0) { Aseg = A0; kloc = k0; } else { Aseg = A1; kloc = k0 - K0; }
        const unsigned short* ga = Aseg + (size_t)(rowBase + rs) * 128 + kloc + cs;
        const unsigned short* gw = Wt + (size_t)rs * Ktot + k0 + cs;
        unsigned short* la = &lA[buf][rs * 32 + cs];
        unsigned short* lw = &lW[buf][rs * 32 + cs];
#if USE_ASYNC_LDS
        __builtin_amdgcn_global_load_async_to_lds_b128(
            (gvec4_t*)ga,       (lvec4_t*)la,       0, 0);
        __builtin_amdgcn_global_load_async_to_lds_b128(
            (gvec4_t*)(ga + 8), (lvec4_t*)(la + 8), 0, 0);
        __builtin_amdgcn_global_load_async_to_lds_b128(
            (gvec4_t*)gw,       (lvec4_t*)lw,       0, 0);
        __builtin_amdgcn_global_load_async_to_lds_b128(
            (gvec4_t*)(gw + 8), (lvec4_t*)(lw + 8), 0, 0);
#else
        *(uint4*)la       = *(const uint4*)ga;
        *(uint4*)(la + 8) = *(const uint4*)(ga + 8);
        *(uint4*)lw       = *(const uint4*)gw;
        *(uint4*)(lw + 8) = *(const uint4*)(gw + 8);
#endif
    };

    stage(0, 0);
    for (int k = 0; k < nk; ++k) {
        const int buf = k & 1;
#if USE_ASYNC_LDS
        __builtin_amdgcn_s_wait_asynccnt(0);   // my chunk-k copies to LDS done
#endif
        __syncthreads();                       // everyone's done; buf^1 is free
        if (k + 1 < nk) stage(k + 1, buf ^ 1); // overlap next copy with compute

        union F { v16bf v; uint4 u[2]; };
        F a, b[8];
        // A 16x32 bf16 fragment (half0: K0-7/16-23, half1: K8-15/24-31)
        a.u[0] = *(const uint4*)&lA[buf][(wave * 16 + m) * 32 + ak];
        a.u[1] = *(const uint4*)&lA[buf][(wave * 16 + m) * 32 + ak + 16];
        #pragma unroll
        for (int t = 0; t < 8; ++t) {
            b[t].u[0] = *(const uint4*)&lW[buf][(t * 16 + m) * 32 + bk];
            b[t].u[1] = *(const uint4*)&lW[buf][(t * 16 + m) * 32 + bk + 16];
        }
        #pragma unroll
        for (int t = 0; t < 8; ++t) {
            acc[t] = __builtin_amdgcn_wmma_f32_16x16x32_bf16(
                         false, a.v, false, b[t].v, (short)0, acc[t], false, false);
        }
    }

    #pragma unroll
    for (int t = 0; t < 8; ++t) {
        float bc = lbias[t * 16 + m];
        #pragma unroll
        for (int r = 0; r < 8; ++r) {
            int row = rowBase + wave * 16 + r + half * 8;  // C/D: M = r + 8*half
            if (row < N) {
                float v = acc[t][r] + bc;
                if (doRelu) v = v > 0.f ? v : 0.f;
                out[(size_t)row * 128 + t * 16 + m] = f2bf(v);
            }
        }
    }
}

// ------------------------------------------------------- edge scatter (COO)
// one wave per edge (uniform e -> scalar loads), 4 channels per lane
__global__ void __launch_bounds__(256)
scatter_kernel(const unsigned short* __restrict__ hw,
               const int* __restrict__ src, const int* __restrict__ dst,
               const float* __restrict__ w, float* __restrict__ agg, int E)
{
    int e = __builtin_amdgcn_readfirstlane(blockIdx.x * 8 + (threadIdx.x >> 5));
    if (e >= E) return;
    int lane = threadIdx.x & 31;
    int s = src[e], d = dst[e];
    float wt = w[e];
    uint2 q = *(const uint2*)&hw[(size_t)s * 128 + lane * 4];
    float* ap = agg + (size_t)d * 128 + lane * 4;
    atomicAdd(ap + 0, bf2f(q.x & 0xFFFFu) * wt);
    atomicAdd(ap + 1, bf2f(q.x >> 16)     * wt);
    atomicAdd(ap + 2, bf2f(q.y & 0xFFFFu) * wt);
    atomicAdd(ap + 3, bf2f(q.y >> 16)     * wt);
}

__global__ void __launch_bounds__(256)
relu_bias_kernel(const float* __restrict__ agg, const float* __restrict__ bias,
                 unsigned short* __restrict__ g, int total)
{
    int i = blockIdx.x * 256 + threadIdx.x;
    if (i < total) {
        float v = agg[i] + bias[i & 127];
        g[i] = f2bf(v > 0.f ? v : 0.f);
    }
}

// ----------------------------------------------------- last2 + sigmoid (f32)
__global__ void __launch_bounds__(128)
final_kernel(const unsigned short* __restrict__ t, const float* __restrict__ W2,
             const float* __restrict__ b2, float* __restrict__ out, int N)
{
    __shared__ float lW[256];
    __shared__ float lb[2];
    int tid = threadIdx.x;
    for (int i = tid; i < 256; i += 128) lW[i] = W2[i];
    if (tid < 2) lb[tid] = b2[tid];
    __syncthreads();
    int n = blockIdx.x * 128 + tid;
    if (n >= N) return;
    float s0 = lb[0], s1 = lb[1];
    #pragma unroll
    for (int c0 = 0; c0 < 128; c0 += 8) {
        uint4 q = *(const uint4*)&t[(size_t)n * 128 + c0];  // 8 bf16 per load
        unsigned int qq[4] = {q.x, q.y, q.z, q.w};
        #pragma unroll
        for (int j = 0; j < 4; ++j) {
            float h0 = bf2f(qq[j] & 0xFFFFu);
            float h1 = bf2f(qq[j] >> 16);
            int c = c0 + j * 2;
            s0 += h0 * lW[c * 2 + 0] + h1 * lW[(c + 1) * 2 + 0];
            s1 += h0 * lW[c * 2 + 1] + h1 * lW[(c + 1) * 2 + 1];
        }
    }
    out[(size_t)n * 2 + 0] = 1.f / (1.f + __expf(-s0));
    out[(size_t)n * 2 + 1] = 1.f / (1.f + __expf(-s1));
}

// ---------------------------------------------------------------------------
extern "C" void kernel_launch(void* const* d_in, const int* in_sizes, int n_in,
                              void* d_out, int out_size, void* d_ws, size_t ws_size,
                              hipStream_t stream)
{
    const float* x    = (const float*)d_in[0];
    const int*   esrc = (const int*)d_in[1];
    const int*   edst = (const int*)d_in[2];
    const float* ew   = (const float*)d_in[3];
    const float* preW = (const float*)d_in[4];
    const float* preB = (const float*)d_in[5];
    const float* f1W  = (const float*)d_in[6];
    const float* f1B  = (const float*)d_in[7];
    const float* f2W  = (const float*)d_in[8];
    const float* f2B  = (const float*)d_in[9];
    const float* gW   = (const float*)d_in[10];
    const float* gB   = (const float*)d_in[11];
    const float* fgW  = (const float*)d_in[12];
    const float* fgB  = (const float*)d_in[13];
    const float* l1W  = (const float*)d_in[14];
    const float* l1B  = (const float*)d_in[15];
    const float* l2W  = (const float*)d_in[16];
    const float* l2B  = (const float*)d_in[17];

    const int N = in_sizes[0] / 16;
    const int E = in_sizes[1];
    const int L = in_sizes[10] / (128 * 128);
    const int nTiles = (N + 127) / 128;
    const int Npad = nTiles * 128;

    char* ws = (char*)d_ws;
    size_t off = 0;
    auto alloc = [&](size_t bytes) -> char* {
        char* p = ws + off;
        off += (bytes + 255) & ~(size_t)255;
        return p;
    };
    unsigned short* node_id = (unsigned short*)alloc((size_t)Npad * 128 * 2);
    unsigned short* hbuf    = (unsigned short*)alloc((size_t)Npad * 128 * 2);
    unsigned short* tmp     = (unsigned short*)alloc((size_t)Npad * 128 * 2); // hw -> g -> t
    float*          agg     = (float*)alloc((size_t)Npad * 128 * 4);
    unsigned short* wg      = (unsigned short*)alloc((size_t)L * 128 * 128 * 2);
    unsigned short* wf      = (unsigned short*)alloc((size_t)L * 256 * 128 * 2);
    unsigned short* wl1     = (unsigned short*)alloc((size_t)256 * 128 * 2);
    (void)ws_size; (void)n_in; (void)out_size;

    const int ngw = L * 128 * 128, nfw = L * 256 * 128, nl1 = 256 * 128;
    cvt_transpose_kernel<<<(ngw + 255) / 256, 256, 0, stream>>>(gW,  wg,  128, ngw);
    cvt_transpose_kernel<<<(nfw + 255) / 256, 256, 0, stream>>>(fgW, wf,  256, nfw);
    cvt_transpose_kernel<<<(nl1 + 255) / 256, 256, 0, stream>>>(l1W, wl1, 256, nl1);

    front_kernel<<<nTiles, 128, 0, stream>>>(x, preW, preB, f1W, f1B, f2W, f2B,
                                             node_id, hbuf, N);

    const int aggN = N * 128;
    for (int l = 0; l < L; ++l) {
        // hw = h @ gcn_W[l]  (no bias, no relu)
        gemm128_kernel<<<nTiles, 256, 0, stream>>>(
            hbuf, nullptr, wg + (size_t)l * 128 * 128, nullptr, tmp, N, 128, 128, 0);
        zero_f32_kernel<<<(aggN + 255) / 256, 256, 0, stream>>>(agg, aggN);
        scatter_kernel<<<(E * 32 + 255) / 256, 256, 0, stream>>>(tmp, esrc, edst, ew, agg, E);
        // g = relu(agg + gcn_b[l])  (g aliases tmp, hw is dead)
        relu_bias_kernel<<<(aggN + 255) / 256, 256, 0, stream>>>(
            agg, gB + (size_t)l * 128, tmp, aggN);
        // h = relu(concat(node_id, g) @ fcg_W[l] + fcg_b[l])
        gemm128_kernel<<<nTiles, 256, 0, stream>>>(
            node_id, tmp, wf + (size_t)l * 256 * 128, fgB + (size_t)l * 128, hbuf,
            N, 128, 256, 1);
    }

    // t = relu(concat(node_id, h) @ last1_W + last1_b)
    gemm128_kernel<<<nTiles, 256, 0, stream>>>(
        node_id, hbuf, wl1, l1B, tmp, N, 128, 256, 1);
    // out = sigmoid(t @ last2_W + last2_b)
    final_kernel<<<nTiles, 128, 0, stream>>>(tmp, l2W, l2B, (float*)d_out, N);
}
